// MGCCDR_74594991997628
// MI455X (gfx1250) — matrix operand order repaired and checked by reading, not compiled, not run on hardware
//
#include <hip/hip_runtime.h>
#include <stdint.h>
#include <stddef.h>

#define D 128
#define C_INV_TEMP 5.0f   // 1 / 0.2

typedef __attribute__((ext_vector_type(16))) _Float16 v16h;
typedef __attribute__((ext_vector_type(8)))  float    v8f;

// ds_swizzle_b32 group-of-32 XOR mode: and_mask=0x1f, or_mask=0, xor_mask=m
#define SWZ_XOR(x, m) \
  __int_as_float(__builtin_amdgcn_ds_swizzle(__float_as_int(x), (0x1f | ((m) << 10))))

static __device__ __forceinline__ float wred32(float x) {  // full-wave sum
  x += SWZ_XOR(x, 1);
  x += SWZ_XOR(x, 2);
  x += SWZ_XOR(x, 4);
  x += SWZ_XOR(x, 8);
  x += SWZ_XOR(x, 16);
  return x;
}

static __device__ __forceinline__ float wred16(float x) {  // sum within 16-lane halves
  x += SWZ_XOR(x, 1);
  x += SWZ_XOR(x, 2);
  x += SWZ_XOR(x, 4);
  x += SWZ_XOR(x, 8);
  return x;
}

static __device__ __forceinline__ float signf(float x) {
  return (x > 0.f) ? 1.f : ((x < 0.f) ? -1.f : 0.f);
}

// Deterministic counter-based uniform in [0,1) (stand-in for threefry noise)
static __device__ __forceinline__ float hash_u01(uint32_t a, uint32_t b, uint32_t c) {
  uint32_t h = a * 0x9E3779B1u + b * 0x85EBCA77u + c * 0xC2B2AE3Du + 0x27D4EB2Fu;
  h ^= h >> 15; h *= 0x2C1B3C6Du;
  h ^= h >> 12; h *= 0x297A2D39u;
  h ^= h >> 15;
  return (float)(h >> 8) * (1.0f / 16777216.0f);
}

// ---------------- elementwise helpers ----------------

__global__ void k_zero(float* __restrict__ p, long n) {
  long i = (long)blockIdx.x * blockDim.x + threadIdx.x;
  if (i < n) p[i] = 0.f;
}

__global__ void k_scale(float* __restrict__ p, long n, float s) {
  long i = (long)blockIdx.x * blockDim.x + threadIdx.x;
  if (i < n) p[i] *= s;
}

// feats = concat(a, b); acc = feats
__global__ void k_concat(const float* __restrict__ a, const float* __restrict__ b,
                         float* __restrict__ feats, float* __restrict__ acc,
                         long na_elems, long n_elems) {
  long i = (long)blockIdx.x * blockDim.x + threadIdx.x;
  if (i >= n_elems) return;
  float v = (i < na_elems) ? a[i] : b[i - na_elems];
  feats[i] = v;
  acc[i]   = v;
}

// ---------------- COO SPMM: y[row] += val * x[col], one wave per edge ----------------

__global__ void k_spmm(const int* __restrict__ row, const int* __restrict__ col,
                       const float* __restrict__ val, const float* __restrict__ x,
                       float* __restrict__ y, int nnz) {
  int e = blockIdx.x * (blockDim.x >> 5) + (threadIdx.x >> 5);
  if (e >= nnz) return;
  int lane = threadIdx.x & 31;
  int r = row[e];
  int c = col[e];
  float v = val[e];
  float4 d = ((const float4*)(x + (long)c * D))[lane];
  float* yr = y + (long)r * D + lane * 4;
  atomicAdd(yr + 0, v * d.x);
  atomicAdd(yr + 1, v * d.y);
  atomicAdd(yr + 2, v * d.z);
  atomicAdd(yr + 3, v * d.w);
}

// ---------------- noise perturb (+ optional normalized accumulate), one wave per row ----------------

template <bool ACCUM>
__global__ void k_noise(float* __restrict__ feats, float* __restrict__ acc,
                        int n, float eps, uint32_t sid, uint32_t layer) {
  int r = blockIdx.x * (blockDim.x >> 5) + (threadIdx.x >> 5);
  if (r >= n) return;
  int lane = threadIdx.x & 31;
  float4* frow = (float4*)(feats + (long)r * D);
  float4 f = frow[lane];
  uint32_t eb = (uint32_t)r * (uint32_t)D + (uint32_t)lane * 4u;
  float4 nz;
  nz.x = hash_u01(sid, layer, eb + 0);
  nz.y = hash_u01(sid, layer, eb + 1);
  nz.z = hash_u01(sid, layer, eb + 2);
  nz.w = hash_u01(sid, layer, eb + 3);
  float nn = wred32(nz.x * nz.x + nz.y * nz.y + nz.z * nz.z + nz.w * nz.w);
  float rs = eps / fmaxf(sqrtf(nn), 1e-12f);
  f.x += signf(f.x) * nz.x * rs;
  f.y += signf(f.y) * nz.y * rs;
  f.z += signf(f.z) * nz.z * rs;
  f.w += signf(f.w) * nz.w * rs;
  frow[lane] = f;
  if (ACCUM) {
    float fn = wred32(f.x * f.x + f.y * f.y + f.z * f.z + f.w * f.w);
    float s = 1.f / fmaxf(sqrtf(fn), 1e-12f);
    float4* arow = (float4*)(acc + (long)r * D);
    float4 a = arow[lane];
    a.x += f.x * s; a.y += f.y * s; a.z += f.z * s; a.w += f.w * s;
    arow[lane] = a;
  }
}

// ---------------- MGA attention merge, one wave per node ----------------

__global__ void k_mga(const float* __restrict__ m_us, const float* __restrict__ m_uv,
                      const float* __restrict__ uv_s, const float* __restrict__ sv_u,
                      const float* __restrict__ m_sv, const float* __restrict__ w,
                      const float* __restrict__ bias, float* __restrict__ rep,
                      int nu, int ntot) {
  int nidx = blockIdx.x * (blockDim.x >> 5) + (threadIdx.x >> 5);
  if (nidx >= ntot) return;
  int lane = threadIdx.x & 31;
  const float4* e0 = (const float4*)(m_us + (long)nidx * D);
  const float4* e1 = (nidx < nu) ? (const float4*)(m_uv + (long)nidx * D)
                                 : (const float4*)(uv_s + (long)(nidx - nu) * D);
  const float4* e2 = (nidx < nu) ? (const float4*)(sv_u + (long)nidx * D)
                                 : (const float4*)(m_sv + (long)(nidx - nu) * D);
  float4 v0 = e0[lane], v1 = e1[lane], v2 = e2[lane];
  float4 wk = ((const float4*)w)[lane];        // lin_w[0:128] applied to key (embs[0])
  float4 wq = ((const float4*)(w + D))[lane];  // lin_w[128:256] applied to query
  float dk = wred32(v0.x * wk.x + v0.y * wk.y + v0.z * wk.z + v0.w * wk.w);
  float q0 = wred32(v0.x * wq.x + v0.y * wq.y + v0.z * wq.z + v0.w * wq.w);
  float q1 = wred32(v1.x * wq.x + v1.y * wq.y + v1.z * wq.z + v1.w * wq.w);
  float q2 = wred32(v2.x * wq.x + v2.y * wq.y + v2.z * wq.z + v2.w * wq.w);
  float b = bias[0];
  float l0 = dk + q0 + b, l1 = dk + q1 + b, l2 = dk + q2 + b;
  float m = fmaxf(l0, fmaxf(l1, l2));
  float x0 = __expf(l0 - m), x1 = __expf(l1 - m), x2 = __expf(l2 - m);
  float inv = 1.f / (x0 + x1 + x2);
  float a0 = x0 * inv, a1 = x1 * inv, a2 = x2 * inv;
  float4 o;
  o.x = a0 * v0.x + a1 * v1.x + a2 * v2.x;
  o.y = a0 * v0.y + a1 * v1.y + a2 * v2.y;
  o.z = a0 * v0.z + a1 * v1.z + a2 * v2.z;
  o.w = a0 * v0.w + a1 * v1.w + a2 * v2.w;
  ((float4*)(rep + (long)nidx * D))[lane] = o;
}

// ---------------- gather + L2-normalize sampled rows; emit f32 + f16 copies ----------------

__global__ void k_gather_norm(const float* __restrict__ rep_base, const int* __restrict__ idx,
                              int idx_stride, float* __restrict__ Pf, _Float16* __restrict__ Ph,
                              float* __restrict__ pos, int Bn) {
  int i = blockIdx.x * (blockDim.x >> 5) + (threadIdx.x >> 5);
  if (i >= Bn) return;
  int lane = threadIdx.x & 31;
  int node = idx[(long)i * idx_stride];
  float4 x = ((const float4*)(rep_base + (long)node * D))[lane];
  float nn = wred32(x.x * x.x + x.y * x.y + x.z * x.z + x.w * x.w);
  float norm = sqrtf(nn);
  float s = 1.f / fmaxf(norm, 1e-12f);
  float4 p; p.x = x.x * s; p.y = x.y * s; p.z = x.z * s; p.w = x.w * s;
  ((float4*)(Pf + (long)i * D))[lane] = p;
  _Float16* ph = Ph + (long)i * D + lane * 4;
  ph[0] = (_Float16)p.x; ph[1] = (_Float16)p.y;
  ph[2] = (_Float16)p.z; ph[3] = (_Float16)p.w;
  if (lane == 0) pos[i] = nn * s * s * C_INV_TEMP;  // dot(p,p)/temp
}

// ---------------- WMMA Gram 16x64 strip + exp row-sums ----------------
// One wave computes a 1x4 strip of 16x16 tiles of S = P @ P^T, reusing the A
// fragments (all four K-chunks) across the strip. |dot| <= 1 so logsumexp
// needs no max subtraction.
//
// A layout (16-bit A 16x32, ISA 7.12.2): lane L: M=L&15; halves 0..7 hold
//   K=koff..koff+7, halves 8..15 hold K=koff+16..koff+23, koff = (L<16)?0:8.
// B layout (B row striped across lanes): lane L: N=L&15; halves t=0..7 hold K=koff+t, t=8..15 hold
//   K = koff+8+t of each 32-wide chunk (koff as for A).  B[k][n] = P[tile_j*16+n][k].
// C/D layout: lane L, VGPR v: M = v + ((L<16)?0:8), N = L&15.

__global__ __launch_bounds__(32) void k_gram_lse(const _Float16* __restrict__ Ph,
                                                 float* __restrict__ rowsum) {
  int ti  = blockIdx.y;
  int tj0 = blockIdx.x * 4;
  int lane = threadIdx.x & 31;
  int MN = lane & 15;
  int hi = (lane >= 16) ? 1 : 0;

  const _Float16* arow = Ph + (long)(ti * 16 + MN) * D + (hi ? 8 : 0);
  v16h A[4];
#pragma unroll
  for (int kc = 0; kc < 4; ++kc) {
#pragma unroll
    for (int h = 0; h < 8; ++h) {
      A[kc][h]     = arow[kc * 32 + h];
      A[kc][8 + h] = arow[kc * 32 + 16 + h];
    }
  }

  int bko = hi ? 8 : 0;
#pragma unroll
  for (int j = 0; j < 4; ++j) {
    const _Float16* brow = Ph + (long)((tj0 + j) * 16 + MN) * D + bko;
    v8f c = {};
#pragma unroll
    for (int kc = 0; kc < 4; ++kc) {
      v16h b;
#pragma unroll
      for (int t = 0; t < 16; ++t) b[t] = brow[kc * 32 + (t < 8 ? t : 8 + t)];
      c = __builtin_amdgcn_wmma_f32_16x16x32_f16(false, A[kc], false, b, (short)0, c,
                                                 false, false);
    }
#pragma unroll
    for (int v = 0; v < 8; ++v) {
      float e = wred16(__expf(c[v] * C_INV_TEMP));  // sum over N (16 lanes per half)
      if (MN == 0) atomicAdd(&rowsum[ti * 16 + v + (hi ? 8 : 0)], e);
    }
  }
}

__global__ void k_closs(const float* __restrict__ rowsum, const float* __restrict__ pos,
                        float* __restrict__ accum, int Bn) {
  int i = blockIdx.x * blockDim.x + threadIdx.x;
  if (i >= Bn) return;
  atomicAdd(accum, logf(rowsum[i]) - pos[i]);  // c_loss = mean(ttl - pos)
}

// ---------------- BPR loss, one wave per sample ----------------

__global__ void k_bpr(const float* __restrict__ rep, const int* __restrict__ users,
                      const int* __restrict__ streamers, float* __restrict__ accum,
                      int nu, int Bn) {
  int i = blockIdx.x * (blockDim.x >> 5) + (threadIdx.x >> 5);
  if (i >= Bn) return;
  int lane = threadIdx.x & 31;
  const float4* u  = (const float4*)(rep + (long)users[i] * D);
  const float4* s0 = (const float4*)(rep + (long)(nu + streamers[2 * i]) * D);
  const float4* s1 = (const float4*)(rep + (long)(nu + streamers[2 * i + 1]) * D);
  float4 uv = u[lane], a = s0[lane], b = s1[lane];
  float d0 = wred32(uv.x * a.x + uv.y * a.y + uv.z * a.z + uv.w * a.w);
  float d1 = wred32(uv.x * b.x + uv.y * b.y + uv.z * b.z + uv.w * b.w);
  if (lane == 0) {
    float diff = d0 - d1;
    // -log_sigmoid(diff) = softplus(-diff), stable form
    float sp = fmaxf(-diff, 0.f) + log1pf(__expf(-fabsf(diff)));
    atomicAdd(accum, sp);
  }
}

__global__ void k_final(const float* __restrict__ acc3, float* __restrict__ out, float invB) {
  out[0] = acc3[0] * invB;                          // bpr
  out[1] = 0.5f * (acc3[1] + acc3[2]) * invB;       // c_loss
}

// ---------------- host orchestration ----------------

static inline int cdiv(long a, long b) { return (int)((a + b - 1) / b); }

extern "C" void kernel_launch(void* const* d_in, const int* in_sizes, int n_in,
                              void* d_out, int out_size, void* d_ws, size_t ws_size,
                              hipStream_t stream) {
  (void)n_in; (void)out_size; (void)ws_size;

  const float* users_feature     = (const float*)d_in[0];
  const float* streamers_feature = (const float*)d_in[1];
  const float* videos_feature    = (const float*)d_in[2];
  const float* lin_w             = (const float*)d_in[3];
  const float* lin_b             = (const float*)d_in[4];
  const int*   us_row = (const int*)d_in[5];
  const int*   us_col = (const int*)d_in[6];
  const float* us_val = (const float*)d_in[7];
  const int*   uv_row = (const int*)d_in[8];
  const int*   uv_col = (const int*)d_in[9];
  const float* uv_val = (const float*)d_in[10];
  const int*   sv_row = (const int*)d_in[11];
  const int*   sv_col = (const int*)d_in[12];
  const float* sv_val = (const float*)d_in[13];
  const int*   uv_agg_row = (const int*)d_in[14];
  const int*   uv_agg_col = (const int*)d_in[15];
  const float* uv_agg_val = (const float*)d_in[16];
  const int*   sv_agg_row = (const int*)d_in[17];
  const int*   sv_agg_col = (const int*)d_in[18];
  const float* sv_agg_val = (const float*)d_in[19];
  const int*   users     = (const int*)d_in[20];
  const int*   streamers = (const int*)d_in[21];

  const int NU = in_sizes[0] / D;
  const int NS = in_sizes[1] / D;
  const int NV = in_sizes[2] / D;
  const int nnz_us  = in_sizes[5];
  const int nnz_uv  = in_sizes[8];
  const int nnz_sv  = in_sizes[11];
  const int nnz_uva = in_sizes[14];
  const int nnz_sva = in_sizes[17];
  const int Bn      = in_sizes[20];   // users is (B,1)

  const int NMAX = (NU + NV > NU + NS) ? ((NU + NV > NS + NV) ? NU + NV : NS + NV)
                                       : ((NU + NS > NS + NV) ? NU + NS : NS + NV);

  // ---- carve workspace ----
  char* wp = (char*)d_ws;
  auto carve = [&](size_t bytes) -> char* {
    char* p = wp;
    wp += (bytes + 255) & ~(size_t)255;
    return p;
  };
  float* P0       = (float*)carve((size_t)NMAX * D * 4);
  float* P1       = (float*)carve((size_t)NMAX * D * 4);
  float* ACC_US   = (float*)carve((size_t)(NU + NS) * D * 4);
  float* ACC_UV   = (float*)carve((size_t)(NU + NV) * D * 4);
  float* ACC_SV   = (float*)carve((size_t)(NS + NV) * D * 4);
  float* UV_S     = (float*)carve((size_t)NS * D * 4);
  float* SV_U     = (float*)carve((size_t)NU * D * 4);
  float* REP      = (float*)carve((size_t)(NU + NS) * D * 4);
  float* PU       = (float*)carve((size_t)Bn * D * 4);
  float* PS       = (float*)carve((size_t)Bn * D * 4);
  _Float16* PHU   = (_Float16*)carve((size_t)Bn * D * 2);
  _Float16* PHS   = (_Float16*)carve((size_t)Bn * D * 2);
  float* POS_U    = (float*)carve((size_t)Bn * 4);
  float* POS_S    = (float*)carve((size_t)Bn * 4);
  float* ROWSUM_U = (float*)carve((size_t)Bn * 4);
  float* ROWSUM_S = (float*)carve((size_t)Bn * 4);
  float* ACC3     = (float*)carve(3 * 4);   // [bpr_sum, closs_u_sum, closs_s_sum]

  const int TB = 256;           // 8 waves/block
  const int WPB = TB / 32;

  // zero scalar accumulators and row-sums
  k_zero<<<1, 32, 0, stream>>>(ACC3, 3);
  k_zero<<<cdiv(Bn, TB), TB, 0, stream>>>(ROWSUM_U, Bn);
  k_zero<<<cdiv(Bn, TB), TB, 0, stream>>>(ROWSUM_S, Bn);

  // ---- propagate helper ----
  auto propagate = [&](const float* a, const float* b, int na, int nb,
                       const int* er, const int* ec, const float* ev, int nnz,
                       float* acc, float eps, uint32_t sid) {
    int n = na + nb;
    long elems = (long)n * D;
    float* f0 = P0;
    float* f1 = P1;
    k_concat<<<cdiv(elems, TB), TB, 0, stream>>>(a, b, f0, acc, (long)na * D, elems);
    for (uint32_t layer = 0; layer < 2; ++layer) {
      k_zero<<<cdiv(elems, TB), TB, 0, stream>>>(f1, elems);
      k_spmm<<<cdiv(nnz, WPB), TB, 0, stream>>>(er, ec, ev, f0, f1, nnz);
      k_noise<true><<<cdiv(n, WPB), TB, 0, stream>>>(f1, acc, n, eps, sid, layer);
      float* t = f0; f0 = f1; f1 = t;
    }
    k_scale<<<cdiv(elems, TB), TB, 0, stream>>>(acc, elems, 1.f / 3.f);
  };

  // ---- aggregate helper ----
  auto aggregate = [&](const int* r, const int* c, const float* v, int nnz,
                       const float* x, float* out, int nout, float eps, uint32_t sid) {
    long elems = (long)nout * D;
    k_zero<<<cdiv(elems, TB), TB, 0, stream>>>(out, elems);
    k_spmm<<<cdiv(nnz, WPB), TB, 0, stream>>>(r, c, v, x, out, nnz);
    k_noise<false><<<cdiv(nout, WPB), TB, 0, stream>>>(out, nullptr, nout, eps, sid, 0u);
  };

  // three propagations (eps = 0.1 each)
  propagate(users_feature, streamers_feature, NU, NS, us_row, us_col, us_val, nnz_us,
            ACC_US, 0.1f, 0u);
  propagate(users_feature, videos_feature, NU, NV, uv_row, uv_col, uv_val, nnz_uv,
            ACC_UV, 0.1f, 1u);
  propagate(streamers_feature, videos_feature, NS, NV, sv_row, sv_col, sv_val, nnz_sv,
            ACC_SV, 0.1f, 2u);

  // UV_s = aggregate(sv_agg, UV_v);  SV_u = aggregate(uv_agg, SV_v)
  aggregate(sv_agg_row, sv_agg_col, sv_agg_val, nnz_sva, ACC_UV + (long)NU * D, UV_S,
            NS, 0.1f, 3u);
  aggregate(uv_agg_row, uv_agg_col, uv_agg_val, nnz_uva, ACC_SV + (long)NS * D, SV_U,
            NU, 0.1f, 4u);

  // MGA attention merge -> REP (users_rep rows [0,NU), streamers_rep rows [NU,NU+NS))
  int ntot = NU + NS;
  k_mga<<<cdiv(ntot, WPB), TB, 0, stream>>>(ACC_US, ACC_UV, UV_S, SV_U, ACC_SV, lin_w,
                                            lin_b, REP, NU, ntot);

  // Gather + normalize sampled rows for contrastive loss
  k_gather_norm<<<cdiv(Bn, WPB), TB, 0, stream>>>(REP, users, 1, PU, PHU, POS_U, Bn);
  k_gather_norm<<<cdiv(Bn, WPB), TB, 0, stream>>>(REP + (long)NU * D, streamers, 2, PS,
                                                  PHS, POS_S, Bn);

  // WMMA Gram + exp row-sums: one wave per 16x64 strip (4 tiles, A reused)
  int gtiles = Bn / 16;                    // 128 for B=2048
  dim3 ggrid((gtiles + 3) / 4, gtiles);
  k_gram_lse<<<ggrid, 32, 0, stream>>>(PHU, ROWSUM_U);
  k_gram_lse<<<ggrid, 32, 0, stream>>>(PHS, ROWSUM_S);

  // c_loss partial sums
  k_closs<<<cdiv(Bn, TB), TB, 0, stream>>>(ROWSUM_U, POS_U, ACC3 + 1, Bn);
  k_closs<<<cdiv(Bn, TB), TB, 0, stream>>>(ROWSUM_S, POS_S, ACC3 + 2, Bn);

  // BPR partial sum
  k_bpr<<<cdiv(Bn, WPB), TB, 0, stream>>>(REP, users, streamers, ACC3, NU, Bn);

  // finalize
  k_final<<<1, 1, 0, stream>>>(ACC3, (float*)d_out, 1.0f / (float)Bn);
}
